// ChunkedSelfAttention_68315749810466
// MI455X (gfx1250) — matrix operand order repaired and checked by reading, not compiled, not run on hardware
//
#include <hip/hip_runtime.h>
#include <hip/hip_bf16.h>
#include <math.h>

// ---------------------------------------------------------------------------
// Chunk-local causal attention with RoPE for MI455X (gfx1250, wave32, WMMA).
// B=4, L=4096, H=16, Dh=Dv=128, CHUNK=1024.
// Flash-attention per (b, chunk, head, 128-row q tile); f16 WMMA, f32 accum.
// ---------------------------------------------------------------------------

typedef __attribute__((ext_vector_type(16))) _Float16 v16h;
typedef __attribute__((ext_vector_type(8)))  _Float16 v8h;
typedef __attribute__((ext_vector_type(8)))  float    v8f;

#define B_SZ   4
#define L_SZ   4096
#define H_SZ   16
#define D_SZ   128
#define CHUNK_SZ 1024
#define BR     128          // q rows per workgroup (8 waves x 16)
#define BC     128          // keys per kv tile
#define LSTR   136          // LDS row stride in f16 (128 + 8 pad, 16B aligned)

#define LOG2E      1.4426950408889634f
#define NEG_BIG    (-1.0e30f)
#define ROPE_COEF  (-0.14391156831212787f)   // -ln(10000)/64
#define QK_SCALE   0.08838834764831843f      // 1/sqrt(128)

static __device__ __forceinline__ v16h cat8h(v8h lo, v8h hi) {
    return __builtin_shufflevector(lo, hi, 0,1,2,3,4,5,6,7,8,9,10,11,12,13,14,15);
}

__global__ __launch_bounds__(256)
void chunked_attn_fa_kernel(const float* __restrict__ q,
                            const float* __restrict__ k,
                            const float* __restrict__ v,
                            const int*   __restrict__ start_index,
                            float*       __restrict__ out) {
    extern __shared__ _Float16 smem[];
    _Float16* lqp = smem;                     // Q tile, later P tile (per-wave rows)
    _Float16* lk  = smem + BR * LSTR;         // K tile, key-major
    _Float16* lv  = smem + 2 * BR * LSTR;     // V tile, dv-major (transposed)

    const int tid  = threadIdx.x;
    const int wave = tid >> 5;
    const int lane = tid & 31;
    const int half = lane >> 4;               // 0/1 lane half
    const int ln   = lane & 15;

    const int bx = blockIdx.x;                // 2048 blocks
    const int t  = bx & 7;                    // q tile within chunk (0..7)
    const int h  = (bx >> 3) & 15;
    const int n  = (bx >> 7) & 3;             // chunk
    const int b  = bx >> 9;
    const int chunk_base = n * CHUNK_SZ;
    const int s0 = start_index[0];

    // ---------------- Stage Q tile (RoPE f32 -> f16) ----------------
    {
        const int q_l0 = chunk_base + t * BR;
        #pragma unroll 1
        for (int it = 0; it < 32; ++it) {
            const int idx = tid + it * 256;       // over 128 rows * 64 pairs
            const int row = idx >> 6;
            const int d   = idx & 63;
            const int l   = q_l0 + row;
            const float* qp = q + (((size_t)b * L_SZ + l) * H_SZ + h) * D_SZ;
            const float x1 = qp[d], x2 = qp[d + 64];
            const float freq = __expf((float)d * ROPE_COEF);
            float cs, sn;
            __sincosf((float)(l + s0) * freq, &cs, &sn);
            lqp[row * LSTR + d]      = (_Float16)(x1 * cs - x2 * sn);
            lqp[row * LSTR + d + 64] = (_Float16)(x2 * cs + x1 * sn);
        }
    }
    __syncthreads();

    // ---------------- Load Q A-fragments (16 rows per wave) ----------------
    // A layout (16x32 f16): lane m=ln holds row (16*wave+ln);
    // elems 0..7 = K 8*half+{0..7}, elems 8..15 = K 16+8*half+{0..7}.
    v16h aq[4];
    {
        const int mrow = 16 * wave + ln;
        #pragma unroll
        for (int s = 0; s < 4; ++s) {
            const _Float16* ap = lqp + mrow * LSTR + 32 * s + 8 * half;
            aq[s] = cat8h(*(const v8h*)ap, *(const v8h*)(ap + 16));
        }
    }

    // ---------------- Flash accumulators ----------------
    v8f o[8];
    float mi[8], li[8];
    #pragma unroll
    for (int c = 0; c < 8; ++c) {
        #pragma unroll
        for (int r = 0; r < 8; ++r) o[c][r] = 0.0f;
    }
    #pragma unroll
    for (int r = 0; r < 8; ++r) { mi[r] = NEG_BIG; li[r] = 0.0f; }

    // ---------------- KV tile loop (causal: kv <= t) ----------------
    for (int kv = 0; kv <= t; ++kv) {
        const int kv_l0 = chunk_base + kv * BR;

        // Stage K tile (RoPE), key-major f16.
        #pragma unroll 1
        for (int it = 0; it < 32; ++it) {
            const int idx = tid + it * 256;
            const int row = idx >> 6;
            const int d   = idx & 63;
            const int l   = kv_l0 + row;
            const float* kp = k + (((size_t)b * L_SZ + l) * H_SZ + h) * D_SZ;
            const float x1 = kp[d], x2 = kp[d + 64];
            const float freq = __expf((float)d * ROPE_COEF);
            float cs, sn;
            __sincosf((float)(l + s0) * freq, &cs, &sn);
            lk[row * LSTR + d]      = (_Float16)(x1 * cs - x2 * sn);
            lk[row * LSTR + d + 64] = (_Float16)(x2 * cs + x1 * sn);
        }
        // Stage V tile transposed: lv[dv * LSTR + key].
        #pragma unroll 1
        for (int it = 0; it < 16; ++it) {
            const int idx = tid + it * 256;       // 128 keys * 32 float4
            const int key = idx >> 5;
            const int d4  = (idx & 31) * 4;
            const float4 vv = *(const float4*)(v + (((size_t)b * L_SZ + kv_l0 + key) * H_SZ + h) * D_SZ + d4);
            lv[(d4 + 0) * LSTR + key] = (_Float16)vv.x;
            lv[(d4 + 1) * LSTR + key] = (_Float16)vv.y;
            lv[(d4 + 2) * LSTR + key] = (_Float16)vv.z;
            lv[(d4 + 3) * LSTR + key] = (_Float16)vv.w;
        }
        __syncthreads();

        // Prefetch next KV tile into cache while computing (global_prefetch_b8).
        if (kv < t) {
            const int nl = chunk_base + (kv + 1) * BR + (tid >> 1);
            const size_t base = (((size_t)b * L_SZ + nl) * H_SZ + h) * D_SZ + (size_t)(tid & 1) * 64;
            __builtin_prefetch(k + base, 0, 0);
            __builtin_prefetch(v + base, 0, 0);
        }

        // ---- S = Q * K^T (f16 WMMA, f32 accum): 8 col tiles x 4 k-steps ----
        // B layout (32x16 f16): lane n=ln is key column; elems 0..7 = K 8*half+{0..7}, elems 8..15 = K 16+8*half+{0..7}.
        v8f sv[8];
        #pragma unroll
        for (int c = 0; c < 8; ++c) {
            v8f acc;
            #pragma unroll
            for (int r = 0; r < 8; ++r) acc[r] = 0.0f;
            const int key = 16 * c + ln;
            #pragma unroll
            for (int s = 0; s < 4; ++s) {
                const _Float16* bp = lk + key * LSTR + 32 * s + 8 * half;
                const v16h bf = cat8h(*(const v8h*)bp, *(const v8h*)(bp + 16));
                acc = __builtin_amdgcn_wmma_f32_16x16x32_f16(
                          false, aq[s], false, bf, (short)0, acc, false, false);
            }
            sv[c] = acc;
        }

        // ---- scale + causal mask + online softmax stats ----
        const bool diag = (kv == t);
        float mnew[8];
        #pragma unroll
        for (int r = 0; r < 8; ++r) mnew[r] = mi[r];
        #pragma unroll
        for (int c = 0; c < 8; ++c) {
            #pragma unroll
            for (int r = 0; r < 8; ++r) {
                float val = sv[c][r] * QK_SCALE;
                if (diag) {
                    const int kcol = 128 * kv + 16 * c + ln;
                    const int qrow = 128 * t + 16 * wave + r + 8 * half;
                    if (kcol > qrow) val = NEG_BIG;
                }
                sv[c][r] = val;
                mnew[r]  = fmaxf(mnew[r], val);
            }
        }
        // row max across 16 lanes of each half (C layout: row spread over N lanes)
        #pragma unroll
        for (int r = 0; r < 8; ++r) {
            mnew[r] = fmaxf(mnew[r], __shfl_xor(mnew[r], 1, 32));
            mnew[r] = fmaxf(mnew[r], __shfl_xor(mnew[r], 2, 32));
            mnew[r] = fmaxf(mnew[r], __shfl_xor(mnew[r], 4, 32));
            mnew[r] = fmaxf(mnew[r], __shfl_xor(mnew[r], 8, 32));
        }
        float alpha[8], psum[8];
        #pragma unroll
        for (int r = 0; r < 8; ++r) {
            alpha[r] = exp2f((mi[r] - mnew[r]) * LOG2E);
            psum[r]  = 0.0f;
            mi[r]    = mnew[r];
        }
        // P = exp(S - m), written to per-wave LDS rows (C layout -> A layout)
        #pragma unroll
        for (int c = 0; c < 8; ++c) {
            #pragma unroll
            for (int r = 0; r < 8; ++r) {
                const float p = exp2f((sv[c][r] - mnew[r]) * LOG2E);
                psum[r] += p;
                lqp[(16 * wave + r + 8 * half) * LSTR + 16 * c + ln] = (_Float16)p;
            }
        }
        #pragma unroll
        for (int r = 0; r < 8; ++r) {
            psum[r] += __shfl_xor(psum[r], 1, 32);
            psum[r] += __shfl_xor(psum[r], 2, 32);
            psum[r] += __shfl_xor(psum[r], 4, 32);
            psum[r] += __shfl_xor(psum[r], 8, 32);
            li[r] = li[r] * alpha[r] + psum[r];
        }
        // rescale O
        #pragma unroll
        for (int c = 0; c < 8; ++c) {
            #pragma unroll
            for (int r = 0; r < 8; ++r) o[c][r] *= alpha[r];
        }

        // ---- load P as A-fragments, O += P * V ----
        v16h ap[4];
        {
            const int mrow = 16 * wave + ln;
            #pragma unroll
            for (int s = 0; s < 4; ++s) {
                const _Float16* pp = lqp + mrow * LSTR + 32 * s + 8 * half;
                ap[s] = cat8h(*(const v8h*)pp, *(const v8h*)(pp + 16));
            }
        }
        #pragma unroll
        for (int c2 = 0; c2 < 8; ++c2) {
            const int dv = 16 * c2 + ln;
            v8f acc = o[c2];
            #pragma unroll
            for (int s = 0; s < 4; ++s) {
                const _Float16* bp = lv + dv * LSTR + 32 * s + 8 * half;
                const v16h bf = cat8h(*(const v8h*)bp, *(const v8h*)(bp + 16));
                acc = __builtin_amdgcn_wmma_f32_16x16x32_f16(
                          false, ap[s], false, bf, (short)0, acc, false, false);
            }
            o[c2] = acc;
        }
        __syncthreads();   // all waves done with lk/lv before restaging
    }

    // ---------------- Epilogue: O / l, store f32 ----------------
    #pragma unroll
    for (int r = 0; r < 8; ++r) {
        const float inv_l = 1.0f / li[r];
        const int l = chunk_base + 128 * t + 16 * wave + r + 8 * half;
        float* op = out + ((size_t)(b * L_SZ + l)) * (H_SZ * D_SZ) + h * D_SZ;
        #pragma unroll
        for (int c2 = 0; c2 < 8; ++c2) {
            op[16 * c2 + ln] = o[c2][r] * inv_l;
        }
    }
}

extern "C" void kernel_launch(void* const* d_in, const int* in_sizes, int n_in,
                              void* d_out, int out_size, void* d_ws, size_t ws_size,
                              hipStream_t stream) {
    (void)in_sizes; (void)n_in; (void)out_size; (void)d_ws; (void)ws_size;
    const float* q  = (const float*)d_in[0];
    const float* k  = (const float*)d_in[1];
    const float* v  = (const float*)d_in[2];
    const int*   si = (const int*)d_in[3];
    float* out = (float*)d_out;

    const dim3 grid(B_SZ * (L_SZ / CHUNK_SZ) * H_SZ * (CHUNK_SZ / BR));  // 2048
    const dim3 block(256);
    const size_t lds_bytes = (size_t)3 * BR * LSTR * sizeof(_Float16);   // ~102 KB
    chunked_attn_fa_kernel<<<grid, block, lds_bytes, stream>>>(q, k, v, si, out);
}
